// MoELayer_61864708931862
// MI455X (gfx1250) — compile-verified, matrix operand order checked by reading, not run on hardware
//
#include <hip/hip_runtime.h>
#include <hip/hip_bf16.h>

// ---------------------------------------------------------------------------
// MoE layer for gfx1250 (MI455X): top-2 routed expert MLP with bf16 WMMA.
// D_MODEL=1024, D_FF=4096, E=8, K=2, T=4096 tokens.
// 512-thread (16-wave) workgroups, 32-token tiles, double-buffered LDS weight
// slabs: one barrier per slab, global-load latency hidden behind WMMA work.
// ---------------------------------------------------------------------------

#define T_TOKENS 4096
#define DM       1024
#define HF       4096
#define NE       8

// expert kernel tiling
#define TM        32      // tokens per tile (M)
#define NTH       512     // threads per block (16 waves)
#define HC        128     // D_FF chunk per iteration
#define KS        64      // K-slab depth staged into LDS per step
#define XS_STRIDE 1032    // 1024+8: 16B-aligned rows, 4-bank row skew
#define WS_STRIDE 72      // 64+8:   16B-aligned rows, 36-bank row skew
#define HT_STRIDE 136     // 128+8:  16B-aligned rows, 4-bank row skew

typedef __attribute__((ext_vector_type(2)))  __bf16 v2bf;
typedef __attribute__((ext_vector_type(4)))  __bf16 v4bf;
typedef __attribute__((ext_vector_type(8)))  __bf16 v8bf;
typedef __attribute__((ext_vector_type(16))) __bf16 v16bf;
typedef __attribute__((ext_vector_type(8)))  float  v8f;

// Load a 16x32 bf16 WMMA fragment (A or B) from a K-contiguous row in LDS.
// Layout (CDNA5 ISA 7.12.2, 16-bit A 16x32): lanes 0-15 hold K=0..7,16..23;
// lanes 16-31 hold K=8..15,24..31; row index = lane%16. Two b128 ds loads.
__device__ inline v16bf ld_frag(const __bf16* row, int k0, int lane) {
    const int b0 = k0 + ((lane & 16) ? 8 : 0);
    v8bf lo = *(const v8bf*)(row + b0);
    v8bf hi = *(const v8bf*)(row + b0 + 16);
    return __builtin_shufflevector(lo, hi, 0, 1, 2, 3, 4, 5, 6, 7,
                                   8, 9, 10, 11, 12, 13, 14, 15);
}

__device__ inline v8f wmma_bf16(v16bf a, v16bf b, v8f c) {
    // (neg_a, A, neg_b, B, c_mod, C, reuse_a, reuse_b)
    return __builtin_amdgcn_wmma_f32_16x16x32_bf16(
        false, a, false, b, (short)0, c, false, false);
}

__device__ inline float fast_tanh(float v) {
#if __has_builtin(__builtin_amdgcn_tanhf)
    return __builtin_amdgcn_tanhf(v);   // gfx1250 v_tanh_f32
#else
    return tanhf(v);
#endif
}

__device__ inline float gelu_tanh(float v) {
    const float k0 = 0.7978845608028654f;  // sqrt(2/pi)
    const float k1 = 0.044715f;
    return 0.5f * v * (1.0f + fast_tanh(k0 * (v + k1 * v * v * v)));
}

// Double-buffered staging of a KS x 128 fp32 weight slab (row stride `rs`
// floats) into a transposed bf16 LDS slab ws[n][k]. The load half issues all
// global_load_b128 back-to-back; the store half packs adjacent k-rows with
// v_cvt_pk_bf16_f32 into ds_store_b32 (LLVM merges to ds_store_2addr_b32).
struct Stage { float4 ra[2], rb[2]; };

__device__ inline void stage_load(const float* __restrict__ gbase, int rs,
                                  int tid, Stage& s) {
#pragma unroll
    for (int i = 0; i < 2; ++i) {       // 1024 (k-pair, n4) groups, 512 thr
        const int g  = tid + i * NTH;
        const int k2 = (g >> 5) << 1;   // even k: 0..62
        const int n4 = (g & 31) << 2;   // 0..124
        s.ra[i] = *(const float4*)(gbase + (size_t)k2 * rs + n4);
        s.rb[i] = *(const float4*)(gbase + (size_t)(k2 + 1) * rs + n4);
    }
}

__device__ inline void stage_store(__bf16 (*__restrict__ ws)[WS_STRIDE],
                                   int tid, const Stage& s) {
#pragma unroll
    for (int i = 0; i < 2; ++i) {
        const int g  = tid + i * NTH;
        const int k2 = (g >> 5) << 1;
        const int n4 = (g & 31) << 2;
        *(v2bf*)&ws[n4 + 0][k2] = v2bf{(__bf16)s.ra[i].x, (__bf16)s.rb[i].x};
        *(v2bf*)&ws[n4 + 1][k2] = v2bf{(__bf16)s.ra[i].y, (__bf16)s.rb[i].y};
        *(v2bf*)&ws[n4 + 2][k2] = v2bf{(__bf16)s.ra[i].z, (__bf16)s.rb[i].z};
        *(v2bf*)&ws[n4 + 3][k2] = v2bf{(__bf16)s.ra[i].w, (__bf16)s.rb[i].w};
    }
}

// ---------------------------------------------------------------------------
// Router: one wave32 per token. Computes 8 logits, softmax, top-2, and
// compacts (token, gate) into per-expert lists with atomics.
// ---------------------------------------------------------------------------
__global__ __launch_bounds__(256) void moe_router_kernel(
    const float* __restrict__ x, const float* __restrict__ Wg,
    int* __restrict__ counts, int* __restrict__ tokids,
    float* __restrict__ gates) {
    const int lane = threadIdx.x & 31;
    const int t = blockIdx.x * 8 + (threadIdx.x >> 5);
    if (t >= T_TOKENS) return;

    const float* xr = x + (size_t)t * DM;
    float logit[NE];
#pragma unroll
    for (int e = 0; e < NE; ++e) {
        float p = 0.f;
        for (int d = lane; d < DM; d += 32)
            p += xr[d] * Wg[d * NE + e];
#pragma unroll
        for (int off = 16; off > 0; off >>= 1)
            p += __shfl_xor(p, off, 32);
        logit[e] = p;  // identical in all lanes after full reduction
    }

    float mx = logit[0];
#pragma unroll
    for (int e = 1; e < NE; ++e) mx = fmaxf(mx, logit[e]);
    float pr[NE];
    float den = 0.f;
#pragma unroll
    for (int e = 0; e < NE; ++e) { pr[e] = __expf(logit[e] - mx); den += pr[e]; }
    const float rden = 1.0f / den;
#pragma unroll
    for (int e = 0; e < NE; ++e) pr[e] *= rden;

    if (lane == 0) {
        int i1 = 0;
#pragma unroll
        for (int e = 1; e < NE; ++e) if (pr[e] > pr[i1]) i1 = e;
        int i2 = (i1 == 0) ? 1 : 0;
#pragma unroll
        for (int e = 0; e < NE; ++e)
            if (e != i1 && pr[e] > pr[i2]) i2 = e;

        int p1 = atomicAdd(&counts[i1], 1);
        tokids[i1 * T_TOKENS + p1] = t;
        gates[i1 * T_TOKENS + p1] = pr[i1];
        int p2 = atomicAdd(&counts[i2], 1);
        tokids[i2 * T_TOKENS + p2] = t;
        gates[i2 * T_TOKENS + p2] = pr[i2];
    }
}

// ---------------------------------------------------------------------------
// Fused expert MLP: grid = (tiles, experts). Each block handles 32 routed
// tokens of one expert: y = gelu(x W1 + b1) W2, accumulated into d_out as
// gate*(y+b2) with f32 global atomics. All GEMM work uses bf16 WMMA; weight
// operands stream through a double-buffered LDS slab (one barrier per slab,
// next slab's global loads issued before the current slab's WMMAs).
// Wave roles (16 waves): mt = wv>>3 selects the 16-token half of the tile,
// nw = wv&7 selects the 16-wide column group within each 128-col chunk.
// ---------------------------------------------------------------------------
__global__ __launch_bounds__(NTH) void moe_expert_kernel(
    const float* __restrict__ x,  const float* __restrict__ W1,
    const float* __restrict__ b1, const float* __restrict__ W2,
    const float* __restrict__ b2, const int* __restrict__ counts,
    const int* __restrict__ tokids, const float* __restrict__ gates,
    float* __restrict__ out) {
    const int e = blockIdx.y;
    const int tile = blockIdx.x;
    const int cnt = counts[e];
    if (tile * TM >= cnt) return;  // inactive tile for this expert

    __shared__ __bf16 xs[TM][XS_STRIDE];        // x tile, bf16, K-contiguous
    __shared__ __bf16 ws2[2][HC][WS_STRIDE];    // double-buffered slab: [n][k]
    __shared__ __bf16 ht[TM][HT_STRIDE];        // gelu(h) chunk: [m][k]
    __shared__ int    s_tok[TM];
    __shared__ float  s_gate[TM];

    const int tid   = threadIdx.x;
    const int lane  = tid & 31;
    const int wv    = tid >> 5;          // 16 waves
    const int mt    = wv >> 3;           // 0/1: which 16-token half
    const int nw    = wv & 7;            // 16-col group within 128-col chunk
    const int lmod  = lane & 15;
    const int khalf = (lane & 16) ? 8 : 0;

    // ---- phase 0: token list + x tile (fp32 -> bf16) ----
    if (tid < TM) {
        const int idx = tile * TM + tid;
        if (idx < cnt) {
            s_tok[tid]  = tokids[(size_t)e * T_TOKENS + idx];
            s_gate[tid] = gates[(size_t)e * T_TOKENS + idx];
        } else {
            s_tok[tid]  = 0;
            s_gate[tid] = 0.f;   // padded row contributes nothing
        }
    }
    __syncthreads();
    {
        float4 r[16];
#pragma unroll
        for (int i = 0; i < 16; ++i) {             // 8192 float4 groups, batched
            const int g   = tid + i * NTH;
            const int row = g >> 8;
            const int c4  = (g & 255) << 2;
            r[i] = *(const float4*)(x + (size_t)s_tok[row] * DM + c4);
        }
#pragma unroll
        for (int i = 0; i < 16; ++i) {
            const int g   = tid + i * NTH;
            const int row = g >> 8;
            const int c4  = (g & 255) << 2;
            *(v4bf*)&xs[row][c4] = v4bf{(__bf16)r[i].x, (__bf16)r[i].y,
                                        (__bf16)r[i].z, (__bf16)r[i].w};
        }
    }
    __syncthreads();

    // persistent GEMM2 accumulators:
    // yacc[c] covers rows mt*16..+16, cols c*128 + nw*16 .. +16
    v8f yacc[8];
#pragma unroll
    for (int c = 0; c < 8; ++c)
        yacc[c] = v8f{0.f, 0.f, 0.f, 0.f, 0.f, 0.f, 0.f, 0.f};

    for (int hc = 0; hc < HF; hc += HC) {
        const float* W1base = W1 + (size_t)e * DM * HF + hc;

        // ---- GEMM1: hacc = x[mt*16..+16,1024] @ W1[1024, hc+nw*16 .. +16) ----
        {   // prologue: stage slab k0=0 into buffer 0
            Stage st;
            stage_load(W1base, HF, tid, st);
            stage_store(ws2[0], tid, st);
        }
        v8f hacc = v8f{0.f, 0.f, 0.f, 0.f, 0.f, 0.f, 0.f, 0.f};
        for (int k0 = 0; k0 < DM; k0 += KS) {
            const int buf = (k0 >> 6) & 1;
            __syncthreads();   // staged slab `buf` visible; buf^1 free
            Stage st;
            const bool nxt = (k0 + KS) < DM;
            if (nxt)  // issue next slab's global loads before the WMMAs
                stage_load(W1base + (size_t)(k0 + KS) * HF, HF, tid, st);
            if (k0 + 2 * KS < DM)  // prefetch slab after next into near caches
                __builtin_prefetch(
                    W1base + (size_t)(k0 + 2 * KS + (tid >> 3)) * HF +
                        ((tid & 7) << 4), 0, 3);
#pragma unroll
            for (int kk = 0; kk < KS; kk += 32) {
                v16bf a = ld_frag(&xs[mt * 16 + lmod][0], k0 + kk, lane);
                v16bf b = ld_frag(&ws2[buf][nw * 16 + lmod][0], kk, lane);
                hacc = wmma_bf16(a, b, hacc);
            }
            if (nxt)
                stage_store(ws2[buf ^ 1], tid, st);
        }

        // bias + gelu -> bf16 h tile in LDS (wave owns rows mt*16.., cols nw*16..)
        {
            const int col = nw * 16 + lmod;
            const float b1v = b1[(size_t)e * HF + hc + col];
#pragma unroll
            for (int r = 0; r < 8; ++r) {
                const int m = mt * 16 + r + khalf;  // C layout row
                ht[m][col] = (__bf16)gelu_tanh(hacc[r] + b1v);
            }
        }
        // no barrier needed here: the first barrier of the GEMM2 stage loop
        // orders ht writes (all waves) before ht reads, and the GEMM2 prologue
        // only writes buffer 0, whose last readers (k0=896) were ordered by the k0=960 barrier.

        // ---- GEMM2: yacc[c] += h[TM, HC] @ W2[hc..hc+HC, c*128 .. +128) ----
        const float* W2base = W2 + ((size_t)e * HF + hc) * DM;
        {   // prologue: stage (k0h=0, c=0) into buffer 0
            Stage st;
            stage_load(W2base, DM, tid, st);
            stage_store(ws2[0], tid, st);
        }
        v16bf a0{}, a1{};
#pragma unroll
        for (int s = 0; s < 16; ++s) {      // s = k0h_idx*8 + c
            const int buf = s & 1;
            __syncthreads();   // staged slab `buf` visible; buf^1 free
            Stage st;
            const bool nxt = (s + 1) < 16;
            if (nxt) {
                const int s1 = s + 1;
                stage_load(W2base + (size_t)((s1 >> 3) * KS) * DM +
                               ((s1 & 7) << 7), DM, tid, st);
            }
            if (s + 2 < 16) {
                const int s2 = s + 2;
                __builtin_prefetch(
                    W2base + (size_t)((s2 >> 3) * KS + (tid >> 3)) * DM +
                        ((s2 & 7) << 7) + ((tid & 7) << 4), 0, 3);
            }
            if ((s & 7) == 0) {  // new k-slab: reload A fragments from ht
                a0 = ld_frag(&ht[mt * 16 + lmod][0], (s >> 3) * KS + 0, lane);
                a1 = ld_frag(&ht[mt * 16 + lmod][0], (s >> 3) * KS + 32, lane);
            }
            const int c = s & 7;
            v16bf b0 = ld_frag(&ws2[buf][nw * 16 + lmod][0], 0, lane);
            yacc[c] = wmma_bf16(a0, b0, yacc[c]);
            v16bf b1f = ld_frag(&ws2[buf][nw * 16 + lmod][0], 32, lane);
            yacc[c] = wmma_bf16(a1, b1f, yacc[c]);
            if (nxt)
                stage_store(ws2[buf ^ 1], tid, st);
        }
        // no trailing barrier: next chunk's GEMM1 prologue writes buffer 0,
        // whose last readers (s=14) were ordered by the s=15 barrier.
    }

    // ---- output: out[t, d] += gate * (y + b2[d]) ----
#pragma unroll
    for (int c = 0; c < 8; ++c) {
        const int colg = c * 128 + nw * 16 + lmod;
        const float b2v = b2[(size_t)e * DM + colg];
#pragma unroll
        for (int r = 0; r < 8; ++r) {
            const int m = mt * 16 + r + khalf;
            const float val = s_gate[m] * (yacc[c][r] + b2v);
            unsafeAtomicAdd(&out[(size_t)s_tok[m] * DM + colg], val);
        }
    }
}

// ---------------------------------------------------------------------------
extern "C" void kernel_launch(void* const* d_in, const int* in_sizes, int n_in,
                              void* d_out, int out_size, void* d_ws, size_t ws_size,
                              hipStream_t stream) {
    (void)in_sizes; (void)n_in; (void)ws_size;
    const float* x  = (const float*)d_in[0];
    const float* Wg = (const float*)d_in[1];
    const float* W1 = (const float*)d_in[2];
    const float* b1 = (const float*)d_in[3];
    const float* W2 = (const float*)d_in[4];
    const float* b2 = (const float*)d_in[5];
    float* out = (float*)d_out;

    // workspace layout: [counts: 8 int (padded to 256B)] [tokids: 8*4096 int]
    //                   [gates: 8*4096 float]
    int*   counts = (int*)d_ws;
    int*   tokids = (int*)((char*)d_ws + 256);
    float* gates  = (float*)((char*)d_ws + 256 + (size_t)NE * T_TOKENS * sizeof(int));

    hipMemsetAsync(d_ws, 0, 256, stream);
    hipMemsetAsync(d_out, 0, (size_t)out_size * sizeof(float), stream);

    moe_router_kernel<<<T_TOKENS / 8, 256, 0, stream>>>(x, Wg, counts, tokids, gates);

    dim3 grid(T_TOKENS / TM, NE);  // worst case: every token routed to one expert
    moe_expert_kernel<<<grid, NTH, 0, stream>>>(x, W1, b1, W2, b2,
                                                counts, tokids, gates, out);
}
